// LSTTaggerBuiltinCell_48996986912982
// MI455X (gfx1250) — matrix operand order repaired and checked by reading, not compiled, not run on hardware
//
#include <hip/hip_runtime.h>
#include <hip/hip_bf16.h>

typedef __attribute__((ext_vector_type(8)))  _Float16 v8h;
typedef __attribute__((ext_vector_type(16))) _Float16 v16h;
typedef __attribute__((ext_vector_type(8)))  float    v8f;
typedef int v4i_ __attribute__((vector_size(16)));   // matches async-LDS builtin param

#define SEQ  4096
#define EMB  512
#define HID  512
#define G4   2048   // 4*HID
#define TAGS 64
#define NWG  32
#define UPW  16     // hidden units per workgroup (recurrent kernel)

#define AS1 __attribute__((address_space(1)))
#define AS3 __attribute__((address_space(3)))

#if __has_builtin(__builtin_amdgcn_global_load_async_to_lds_b128)
#define USE_ASYNC_LDS 1
#else
#define USE_ASYNC_LDS 0
#endif

static __device__ __forceinline__ void wait_async0() {
#if __has_builtin(__builtin_amdgcn_s_wait_asynccnt)
    __builtin_amdgcn_s_wait_asynccnt(0);
#else
    asm volatile("s_wait_asynccnt 0x0" ::: "memory");
#endif
}

// ---------------------------------------------------------------- utilities
__global__ void init_sync_kernel(unsigned* cnt) { *cnt = 0u; }

__global__ void cast_f32_to_f16(const float* __restrict__ src,
                                _Float16* __restrict__ dst, int n) {
    int i = blockIdx.x * blockDim.x + threadIdx.x;
    if (i < n) dst[i] = (_Float16)src[i];
}

__global__ void gather_cast_kernel(const int* __restrict__ seq,
                                   const float* __restrict__ emb,
                                   _Float16* __restrict__ xs) {
    int i = blockIdx.x * blockDim.x + threadIdx.x;   // over SEQ*EMB
    int t = i >> 9;                                  // EMB == 512
    int e = i & 511;
    xs[i] = (_Float16)emb[(size_t)seq[t] * EMB + e];
}

// ------------------------------------------------- Xproj = xs @ W_ih^T + b
// One wave computes a 16(M) x 64(N) tile: 4 WMMA accumulators, K swept 512.
// A-frag: lane l holds row (l&15), K = (l>>4)*16..+15 (contiguous in xs row).
// B-frag: lane l holds col (l&15), K = (l>>4)*16..+15 (contiguous in W_ih row).
__global__ void __launch_bounds__(256)
gemm_xproj_kernel(const _Float16* __restrict__ xs,   // [SEQ, EMB] f16
                  const _Float16* __restrict__ wih,  // [G4, EMB]  f16 row-major
                  const float* __restrict__ b_ih,
                  const float* __restrict__ b_hh,
                  float* __restrict__ xp)            // [SEQ, G4]  f32
{
    const int lane  = threadIdx.x & 31;
    const int wid   = blockIdx.x * 8 + (threadIdx.x >> 5);  // 0..8191
    const int nGrps = G4 / 64;                              // 32
    const int m0    = (wid / nGrps) * 16;                   // 0..4080
    const int n0    = (wid % nGrps) * 64;
    const int rsel  = lane & 15;
    const int khalf = lane >> 4;

    v8f acc[4];
#pragma unroll
    for (int j = 0; j < 4; ++j) {
        const int c = n0 + j * 16 + rsel;
        const float bv = b_ih[c] + b_hh[c];   // bias depends only on N
        v8f t;
#pragma unroll
        for (int r = 0; r < 8; ++r) t[r] = bv;
        acc[j] = t;
    }

    const _Float16* aRow = xs + (size_t)(m0 + rsel) * EMB;
#pragma unroll 4
    for (int k0 = 0; k0 < EMB; k0 += 32) {
        v8h alo = *(const v8h*)(aRow + k0 + khalf * 16);
        v8h ahi = *(const v8h*)(aRow + k0 + khalf * 16 + 8);
        v16h a;
#pragma unroll
        for (int q = 0; q < 8; ++q) { a[q] = alo[q]; a[8 + q] = ahi[q]; }
#pragma unroll
        for (int j = 0; j < 4; ++j) {
            const _Float16* bp =
                wih + (size_t)(n0 + j * 16 + rsel) * EMB + k0 + khalf * 16;
            v16h b = *(const v16h*)bp;
            acc[j] = __builtin_amdgcn_wmma_f32_16x16x32_f16(
                false, a, false, b, (short)0, acc[j], false, false);
        }
    }

    // D layout: lane l holds N = n0+j*16+(l&15); rows r + 8*(l>>4).
#pragma unroll
    for (int j = 0; j < 4; ++j)
#pragma unroll
        for (int r = 0; r < 8; ++r)
            xp[(size_t)(m0 + r + 8 * khalf) * G4 + n0 + j * 16 + rsel] =
                acc[j][r];
}

// --------------------------------------------------------- LSTM recurrence
// 32 WGs x 256 threads; WG owns 16 hidden units (64 gate dots of length 512).
// CDNA5 WGP has 320 KB LDS: the WG's whole W_hh slice (64 KB f16) is staged
// in LDS once, so per-step dot products never touch L2. The per-step h
// broadcast refresh and the one-time W fill use GLOBAL_LOAD_ASYNC_TO_LDS
// (ASYNCcnt) when available: memory->LDS with no VGPR round trip, shortening
// the serial critical path.
__global__ void __launch_bounds__(256)
lstm_recurrent_kernel(const _Float16* __restrict__ whh,  // [G4, HID] f16
                      const float* __restrict__ xp,      // [SEQ, G4]
                      float* __restrict__ hg,            // [HID] shared h
                      unsigned* __restrict__ cnt,
                      unsigned* __restrict__ gen)
{
    __shared__ __align__(16) _Float16 sW[64 * HID];  // 64 KB W_hh slice
    __shared__ __align__(16) float sh[HID];
    __shared__ float sc[UPW];
    __shared__ float sg[64];
    __shared__ float sx[64];
    const int tid  = threadIdx.x;
    const int wg   = blockIdx.x;
    const int wave = tid >> 5;
    const int lane = tid & 31;

    // one-time LDS fill: sW row d(=gate*16+u) <- whh[gate*512 + wg*16 + u]
    for (int c = tid; c < 64 * 64; c += 256) {       // 4096 x 16B chunks
        const int d    = c >> 6;                     // 0..63
        const int q    = c & 63;                     // uint4 within row
        const int gate = d >> 4, u = d & 15;
        const uint4* src =
            (const uint4*)(whh + (size_t)(gate * HID + wg * UPW + u) * HID) + q;
        uint4* dst = (uint4*)sW + c;
#if USE_ASYNC_LDS
        __builtin_amdgcn_global_load_async_to_lds_b128(
            (AS1 v4i_*)(void*)src, (AS3 v4i_*)(void*)dst, 0, 0);
#else
        *dst = *src;
#endif
    }
#if USE_ASYNC_LDS
    wait_async0();
#endif
    for (int i = tid; i < HID; i += 256) sh[i] = 0.f;
    if (tid < UPW) sc[tid] = 0.f;
    __syncthreads();

    for (int t = 0; t < SEQ; ++t) {
        // prefetch this step's input-projection gates (stays in flight)
        float xpv = 0.f;
        if (tid < 64) {
            const int gate = tid >> 4, u = tid & 15;
            xpv = xp[(size_t)t * G4 + gate * HID + wg * UPW + u];
        }

        // each lane caches its h chunk once, reused across 8 dots
        float hreg[16];
#pragma unroll
        for (int q = 0; q < 16; ++q) hreg[q] = sh[lane * 16 + q];

#pragma unroll
        for (int q8 = 0; q8 < 8; ++q8) {
            const int d = wave * 8 + q8;             // 0..63
            const _Float16* wr = sW + d * HID + lane * 16;
            v8h w0 = *(const v8h*)(wr);
            v8h w1 = *(const v8h*)(wr + 8);
            float acc = 0.f;
#pragma unroll
            for (int q = 0; q < 8; ++q) acc += (float)w0[q] * hreg[q];
#pragma unroll
            for (int q = 0; q < 8; ++q) acc += (float)w1[q] * hreg[8 + q];
#pragma unroll
            for (int m = 16; m >= 1; m >>= 1) acc += __shfl_xor(acc, m, 32);
            if (lane == 0) sg[d] = acc;
        }
        if (tid < 64) sx[tid] = xpv;
        __syncthreads();

        if (tid < UPW) {
            const int u = tid;
            float iv = sg[u]      + sx[u];
            float fv = sg[16 + u] + sx[16 + u];
            float gv = sg[32 + u] + sx[32 + u];
            float ov = sg[48 + u] + sx[48 + u];
            iv = 1.f / (1.f + __expf(-iv));
            fv = 1.f / (1.f + __expf(-fv));
            gv = tanhf(gv);
            ov = 1.f / (1.f + __expf(-ov));
            const float c = fv * sc[u] + iv * gv;
            sc[u] = c;
            hg[wg * UPW + u] = ov * tanhf(c);
        }
        __threadfence();      // release h slice to agent scope
        __syncthreads();

        // grid-wide sense-reversing barrier (L2 atomics)
        if (tid == 0) {
            unsigned g = __hip_atomic_load(gen, __ATOMIC_RELAXED,
                                           __HIP_MEMORY_SCOPE_AGENT);
            unsigned a = __hip_atomic_fetch_add(cnt, 1u, __ATOMIC_ACQ_REL,
                                                __HIP_MEMORY_SCOPE_AGENT);
            if (a == NWG - 1) {
                __hip_atomic_store(cnt, 0u, __ATOMIC_RELAXED,
                                   __HIP_MEMORY_SCOPE_AGENT);
                __hip_atomic_store(gen, g + 1u, __ATOMIC_RELEASE,
                                   __HIP_MEMORY_SCOPE_AGENT);
            } else {
                while (__hip_atomic_load(gen, __ATOMIC_ACQUIRE,
                                         __HIP_MEMORY_SCOPE_AGENT) == g)
                    __builtin_amdgcn_s_sleep(2);
            }
        }
        __builtin_amdgcn_s_cluster_barrier(); // HW fast path if cluster-dispatched; NOP otherwise
        __syncthreads();
        __threadfence();      // acquire: invalidate L0 before re-reading h

        // refresh the shared-h copy for the next step
#if USE_ASYNC_LDS
        if (tid < HID / 4) {  // 128 lanes x 16B = 2 KB, straight to LDS
            __builtin_amdgcn_global_load_async_to_lds_b128(
                (AS1 v4i_*)(void*)(hg + tid * 4),
                (AS3 v4i_*)(void*)(sh + tid * 4), 0, 0);
        }
        wait_async0();
#else
        for (int i = tid; i < HID; i += 256) sh[i] = hg[i];
#endif
        __syncthreads();
    }
}

// ------------------------------------------------ logits + log_softmax [64]
__global__ void __launch_bounds__(256)
output_logsoftmax_kernel(const float* __restrict__ hg,
                         const float* __restrict__ wout,  // [TAGS, HID]
                         const float* __restrict__ bout,
                         float* __restrict__ out)
{
    __shared__ float partial[256];
    __shared__ float logits[TAGS];
    const int tid  = threadIdx.x;
    const int d    = tid >> 2;   // 0..63
    const int part = tid & 3;
    const float* wr = wout + (size_t)d * HID + part * 128;
    const float* hr = hg + part * 128;
    float acc = 0.f;
    for (int q = 0; q < 128; ++q) acc += wr[q] * hr[q];
    partial[tid] = acc;
    __syncthreads();
    if (part == 0)
        logits[d] = partial[tid] + partial[tid + 1] + partial[tid + 2] +
                    partial[tid + 3] + bout[d];
    __syncthreads();
    if (tid == 0) {
        float mx = -1e30f;
        for (int i = 0; i < TAGS; ++i) mx = fmaxf(mx, logits[i]);
        float s = 0.f;
        for (int i = 0; i < TAGS; ++i) s += __expf(logits[i] - mx);
        const float lse = mx + logf(s);
        for (int i = 0; i < TAGS; ++i) out[i] = logits[i] - lse;
    }
}

// ---------------------------------------------------------------- launcher
extern "C" void kernel_launch(void* const* d_in, const int* in_sizes, int n_in,
                              void* d_out, int out_size, void* d_ws, size_t ws_size,
                              hipStream_t stream) {
    const int*   seq   = (const int*)  d_in[0];
    const float* emb   = (const float*)d_in[1];
    const float* W_ih  = (const float*)d_in[2];
    const float* W_hh  = (const float*)d_in[3];
    const float* b_ih  = (const float*)d_in[4];
    const float* b_hh  = (const float*)d_in[5];
    const float* W_out = (const float*)d_in[6];
    const float* b_out = (const float*)d_in[7];
    float* out = (float*)d_out;

    char* ws = (char*)d_ws;
    _Float16* xs_h  = (_Float16*)(ws);                       //  4 MB
    _Float16* wih_h = (_Float16*)(ws + (4u  << 20));         //  2 MB
    _Float16* whh_h = (_Float16*)(ws + (6u  << 20));         //  2 MB
    float*    xp    = (float*)   (ws + (8u  << 20));         // 32 MB
    float*    hg    = (float*)   (ws + (40u << 20));         //  2 KB
    unsigned* cnt   = (unsigned*)(ws + (40u << 20) + 4096);
    unsigned* gen   = cnt + 1;

    init_sync_kernel<<<1, 1, 0, stream>>>(cnt);

    cast_f32_to_f16<<<(G4 * EMB) / 256, 256, 0, stream>>>(W_ih, wih_h, G4 * EMB);
    cast_f32_to_f16<<<(G4 * HID) / 256, 256, 0, stream>>>(W_hh, whh_h, G4 * HID);
    gather_cast_kernel<<<(SEQ * EMB) / 256, 256, 0, stream>>>(seq, emb, xs_h);

    gemm_xproj_kernel<<<1024, 256, 0, stream>>>(xs_h, wih_h, b_ih, b_hh, xp);

    lstm_recurrent_kernel<<<NWG, 256, 0, stream>>>(whh_h, xp, hg, cnt, gen);

    output_logsoftmax_kernel<<<1, 256, 0, stream>>>(hg, W_out, b_out, out);
}
